// MCAttention_81329500717213
// MI455X (gfx1250) — compile-verified, matrix operand order checked by reading, not run on hardware
//
#include <hip/hip_runtime.h>
#include <hip/hip_bf16.h>
#include <math.h>

// ---------------------------------------------------------------------------
// MCAttention for MI455X (gfx1250, wave32, WMMA f32_16x16x32_f16)
//
// Shapes: E=1024, H=16, B=8, HD=64, LA=LV=512, LL=128, S=1152.
// Pipeline:
//   1) init_scales : per-key-column scale (count^-0.5) and -inf pad bias
//   2) wmma_gemm_bias x3 : qkv = x @ W^T + b          (M,3072,1024)
//   3) mhca_attn   x3 : scores->softmax->map, P@V     (single-pass via big LDS)
//   4) wmma_gemm_bias x3 : out = attn @ Wo^T + bo     (M,1024,1024)
// ---------------------------------------------------------------------------

typedef __attribute__((ext_vector_type(16))) _Float16 v16h;
typedef __attribute__((ext_vector_type(8)))  float    v8f;

__device__ __forceinline__ v8f zero8() {
  v8f z = {0.f, 0.f, 0.f, 0.f, 0.f, 0.f, 0.f, 0.f};
  return z;
}

// Build a 16x32 f16 A/B fragment from two contiguous 8-float runs (with scale).
// Lane layout (ISA 16-bit A-matrix 16x32): halves [0:8]=K kbase..kbase+7,
// halves [8:16]=K 16+kbase..16+kbase+7, kbase = 8*(lane>>4), row = lane&15.
__device__ __forceinline__ v16h frag_from_f32(const float* p0, const float* p1,
                                              float scale) {
  float a[8], b[8];
  __builtin_memcpy(a, p0, 32);
  __builtin_memcpy(b, p1, 32);
  v16h f;
#pragma unroll
  for (int i = 0; i < 8; ++i) {
    f[i]     = (_Float16)(a[i] * scale);
    f[i + 8] = (_Float16)(b[i] * scale);
  }
  return f;
}

// Same, but from two contiguous 8-half runs in LDS.
__device__ __forceinline__ v16h frag_from_f16(const _Float16* p0,
                                              const _Float16* p1) {
  v16h f;
  __builtin_memcpy(&f, p0, 16);
  __builtin_memcpy((char*)&f + 16, p1, 16);
  return f;
}

__device__ __forceinline__ const float* key_base(const float* qa,
                                                 const float* qv,
                                                 const float* ql, int s,
                                                 int* ls) {
  if (s < 512)  { *ls = s;        return qa; }
  if (s < 1024) { *ls = s - 512;  return qv; }
  *ls = s - 1024; return ql;
}

// ---------------------------------------------------------------------------
// Generic GEMM:  C[M,N] = A[M,K](f32) * W[N,K](f32)^T + bias[N]
// Block tile 128x128, BK=32, double-buffered LDS with register staging.
// 8 waves as 4(M)x2(N); each wave owns a 32x64 tile = 2x4 WMMA tiles,
// i.e. 8 v_wmma per K-slice with a single barrier per slice and the next
// slice's global loads in flight during the WMMAs.
// ---------------------------------------------------------------------------
#define GLDA 40  // padded LDS row stride in halves (32 + 8)

__global__ __launch_bounds__(256) void wmma_gemm_bias(
    const float* __restrict__ A, const float* __restrict__ W,
    const float* __restrict__ bias, float* __restrict__ C, int M, int N,
    int K) {
  __shared__ __align__(16) _Float16 As[2][128 * GLDA];
  __shared__ __align__(16) _Float16 Bs[2][128 * GLDA];

  const int tid = threadIdx.x;
  const int lane = tid & 31, wave = tid >> 5;
  const int waveM = wave & 3, waveN = wave >> 2;
  const int m0 = waveM * 32, n0 = waveN * 64;
  const int lk = lane & 15, koff = (lane >> 4) * 8, rhi = (lane >> 4) * 8;
  const int mBase = blockIdx.y * 128, nBase = blockIdx.x * 128;

  // Per-thread staging slots: 4 float4 of A + 4 float4 of B per K-slice.
  const int ldR = tid >> 3;          // 0..31 row group base
  const int ldC = (tid & 7) * 4;     // column within 32-wide slice

  v8f acc[2][4];
#pragma unroll
  for (int mt = 0; mt < 2; ++mt)
#pragma unroll
    for (int nt = 0; nt < 4; ++nt) acc[mt][nt] = zero8();

  float4 aS[4], bS[4];

  auto load_regs = [&](int kb) {
#pragma unroll
    for (int i = 0; i < 4; ++i) {
      int r = ldR + i * 32;
      aS[i] = *(const float4*)(A + (size_t)(mBase + r) * K + kb + ldC);
      bS[i] = *(const float4*)(W + (size_t)(nBase + r) * K + kb + ldC);
    }
  };
  auto store_lds = [&](int buf) {
#pragma unroll
    for (int i = 0; i < 4; ++i) {
      int r = ldR + i * 32;
      _Float16* da = &As[buf][r * GLDA + ldC];
      da[0] = (_Float16)aS[i].x; da[1] = (_Float16)aS[i].y;
      da[2] = (_Float16)aS[i].z; da[3] = (_Float16)aS[i].w;
      _Float16* db = &Bs[buf][r * GLDA + ldC];
      db[0] = (_Float16)bS[i].x; db[1] = (_Float16)bS[i].y;
      db[2] = (_Float16)bS[i].z; db[3] = (_Float16)bS[i].w;
    }
  };

  const int steps = K >> 5;
  load_regs(0);
  store_lds(0);
  __syncthreads();

  for (int k = 0; k < steps; ++k) {
    const int cur = k & 1;
    if (k + 1 < steps) load_regs((k + 1) << 5);

    v16h aF[2], bF[4];
#pragma unroll
    for (int mt = 0; mt < 2; ++mt) {
      const _Float16* p = &As[cur][(m0 + mt * 16 + lk) * GLDA + koff];
      aF[mt] = frag_from_f16(p, p + 16);
    }
#pragma unroll
    for (int nt = 0; nt < 4; ++nt) {
      const _Float16* p = &Bs[cur][(n0 + nt * 16 + lk) * GLDA + koff];
      bF[nt] = frag_from_f16(p, p + 16);
    }
#pragma unroll
    for (int mt = 0; mt < 2; ++mt)
#pragma unroll
      for (int nt = 0; nt < 4; ++nt)
        acc[mt][nt] = __builtin_amdgcn_wmma_f32_16x16x32_f16(
            false, aF[mt], false, bF[nt], (short)0, acc[mt][nt], false, false);

    if (k + 1 < steps) {
      store_lds((k + 1) & 1);
      __syncthreads();
    }
  }

  float bn[4];
#pragma unroll
  for (int nt = 0; nt < 4; ++nt) bn[nt] = bias[nBase + n0 + nt * 16 + lk];
#pragma unroll
  for (int mt = 0; mt < 2; ++mt) {
#pragma unroll
    for (int r = 0; r < 8; ++r) {
      size_t row = (size_t)(mBase + m0 + mt * 16 + rhi + r) * N;
#pragma unroll
      for (int nt = 0; nt < 4; ++nt)
        C[row + nBase + n0 + nt * 16 + lk] = acc[mt][nt][r] + bn[nt];
    }
  }
}

// ---------------------------------------------------------------------------
// Per-key-column scale mask / additive mask precompute.
// kscale[b][s] = (#unmasked in s's modality, batch b)^-0.5
// kbias [b][s] = masked ? -inf : 0
// ---------------------------------------------------------------------------
__global__ void init_scales(const unsigned char* __restrict__ ma,
                            const unsigned char* __restrict__ mv,
                            const unsigned char* __restrict__ ml,
                            float* __restrict__ kscale,
                            float* __restrict__ kbias) {
  __shared__ float sc3[24];  // [modality*8 + b]
  int tid = threadIdx.x;
  if (tid < 24) {
    int m = tid >> 3, b = tid & 7;
    const unsigned char* mp = (m == 0) ? ma : ((m == 1) ? mv : ml);
    int L = (m == 2) ? 128 : 512;
    int cnt = 0;
    for (int l = 0; l < L; ++l) cnt += (mp[b * L + l] == 0) ? 1 : 0;
    sc3[tid] = rsqrtf((float)cnt);
  }
  __syncthreads();
  for (int i = tid; i < 8 * 1152; i += blockDim.x) {
    int b = i / 1152, s = i % 1152;
    int m = (s < 512) ? 0 : ((s < 1024) ? 1 : 2);
    int ls = s - ((m == 0) ? 0 : ((m == 1) ? 512 : 1024));
    int L = (m == 2) ? 128 : 512;
    const unsigned char* mp = (m == 0) ? ma : ((m == 1) ? mv : ml);
    bool masked = mp[b * L + ls] != 0;
    kscale[i] = sc3[m * 8 + b];
    kbias[i] = masked ? -INFINITY : 0.0f;
  }
}

// ---------------------------------------------------------------------------
// Cross-modal attention for one query modality.
// Grid: (Tq/16, B*H). Block: 256 threads (8 waves).
// One block owns a 16-query strip for one (b,h): the full 16x1152 score row
// lives in LDS (CDNA5 has 320KB/WGP), enabling single-pass
// scores -> softmax -> map write -> P@V. The per-wave V-staging LDS region is
// reused as the f32 partial buffer for the cross-wave output reduction, so the
// final output is written with plain stores (tile is block-exclusive).
// ---------------------------------------------------------------------------
#define S_ALL 1152
#define SC_STRIDE 1160  // padded f32 row stride (16B-aligned rows)
#define VT_LD 40        // padded Vt row stride in halves

__global__ __launch_bounds__(256) void mhca_attn(
    const float* __restrict__ qkv_q,   // query modality (Tq*B, 3E)
    const float* __restrict__ qkv_ka,  // audio qkv (512*B, 3E)
    const float* __restrict__ qkv_kv,  // vision qkv
    const float* __restrict__ qkv_kl,  // text qkv
    const float* __restrict__ kscale, const float* __restrict__ kbias,
    float* __restrict__ attn_buf,  // (Tq*B, E)
    float* __restrict__ map_out,   // (B, Tq, S), pre-zeroed
    int Tq) {
  extern __shared__ __align__(16) unsigned char smem[];
  float* sc = (float*)smem;                               // 16 x 1160 f32
  _Float16* Vt = (_Float16*)(smem + 16 * SC_STRIDE * 4);  // 8 x 64 x 40 f16
  float* red = (float*)(smem + 16 * SC_STRIDE * 4 + 8 * 64 * VT_LD * 2);
  float* rowmax = red + 256;
  float* rowsum = rowmax + 16;

  const int tid = threadIdx.x, lane = tid & 31, wave = tid >> 5;
  const int lk = lane & 15, koff = (lane >> 4) * 8, rhi = (lane >> 4) * 8;
  const int bh = blockIdx.y;
  const int b = bh >> 4, h = bh & 15;
  const int t0 = blockIdx.x * 16;

  // ---- Q fragments in registers (pre-scaled by HD^-0.5 = 0.125) ----
  const float* qrow = qkv_q + (size_t)((t0 + lk) * 8 + b) * 3072 + h * 64;
  v16h q0 = frag_from_f32(qrow + koff, qrow + 16 + koff, 0.125f);
  v16h q1 = frag_from_f32(qrow + 32 + koff, qrow + 48 + koff, 0.125f);

  // ---- Phase 1: scores = (Q*s) . K^T, scaled + masked, into LDS strip ----
  for (int st = wave * 16; st < S_ALL; st += 128) {
    int ls;
    const float* kp = key_base(qkv_ka, qkv_kv, qkv_kl, st, &ls);
    const float* krow = kp + (size_t)((ls + lk) * 8 + b) * 3072 + 1024 + h * 64;
    v16h k0 = frag_from_f32(krow + koff, krow + 16 + koff, 1.0f);
    v16h k1 = frag_from_f32(krow + 32 + koff, krow + 48 + koff, 1.0f);
    v8f s_acc = zero8();
    s_acc = __builtin_amdgcn_wmma_f32_16x16x32_f16(false, q0, false, k0,
                                                   (short)0, s_acc, false, false);
    s_acc = __builtin_amdgcn_wmma_f32_16x16x32_f16(false, q1, false, k1,
                                                   (short)0, s_acc, false, false);
    int sg = st + lk;
    float ksc = kscale[b * S_ALL + sg];
    float kbi = kbias[b * S_ALL + sg];
#pragma unroll
    for (int r = 0; r < 8; ++r)
      sc[(rhi + r) * SC_STRIDE + sg] = s_acc[r] * ksc + kbi;
  }
  __syncthreads();

  // ---- Phase 2: softmax over S per query row; emit head-mean map ----
  {
    int row = tid >> 4, c0 = tid & 15;
    float m = -3.4e38f;
    for (int s = c0; s < S_ALL; s += 16)
      m = fmaxf(m, sc[row * SC_STRIDE + s]);
    red[row * 16 + c0] = m;
    __syncthreads();
    if (c0 == 0) {
      float mm = -3.4e38f;
#pragma unroll
      for (int i = 0; i < 16; ++i) mm = fmaxf(mm, red[row * 16 + i]);
      rowmax[row] = mm;
    }
    __syncthreads();
    float rm = rowmax[row];
    float sum = 0.f;
    for (int s = c0; s < S_ALL; s += 16) {
      float e = __expf(sc[row * SC_STRIDE + s] - rm);
      sc[row * SC_STRIDE + s] = e;
      sum += e;
    }
    red[row * 16 + c0] = sum;
    __syncthreads();
    if (c0 == 0) {
      float t = 0.f;
#pragma unroll
      for (int i = 0; i < 16; ++i) t += red[row * 16 + i];
      rowsum[row] = t;
    }
    __syncthreads();
    float inv = 1.0f / rowsum[row];
    for (int s = c0; s < S_ALL; s += 16) {
      float p = sc[row * SC_STRIDE + s] * inv;
      sc[row * SC_STRIDE + s] = p;  // keep probs for P@V
      atomicAdd(&map_out[((size_t)b * Tq + t0 + row) * S_ALL + s],
                p * 0.0625f);  // mean over H=16 heads
    }
    __syncthreads();
  }

  // ---- Phase 3: out = P @ V, S split across waves in chunks of 32 ----
  v8f o0 = zero8(), o1 = zero8(), o2 = zero8(), o3 = zero8();
  _Float16* vtw = Vt + wave * (64 * VT_LD);
  for (int c = wave; c < S_ALL / 32; c += 8) {
    int s0 = c * 32;
    int ls;
    const float* vp = key_base(qkv_ka, qkv_kv, qkv_kl, s0, &ls);
    // Stage this wave's 32-key V chunk transposed (d-major) into LDS as f16.
    const float* vrow = vp + (size_t)((ls + lane) * 8 + b) * 3072 + 2048 + h * 64;
#pragma unroll
    for (int d4 = 0; d4 < 16; ++d4) {
      float4 v = *(const float4*)(vrow + d4 * 4);
      vtw[(d4 * 4 + 0) * VT_LD + lane] = (_Float16)v.x;
      vtw[(d4 * 4 + 1) * VT_LD + lane] = (_Float16)v.y;
      vtw[(d4 * 4 + 2) * VT_LD + lane] = (_Float16)v.z;
      vtw[(d4 * 4 + 3) * VT_LD + lane] = (_Float16)v.w;
    }
    asm volatile("s_wait_dscnt 0" ::: "memory");  // intra-wave LDS visibility

    const float* pp = sc + lk * SC_STRIDE + s0 + koff;
    v16h pa = frag_from_f32(pp, pp + 16, 1.0f);
    const _Float16* v0p = vtw + (0 * 16 + lk) * VT_LD + koff;
    const _Float16* v1p = vtw + (1 * 16 + lk) * VT_LD + koff;
    const _Float16* v2p = vtw + (2 * 16 + lk) * VT_LD + koff;
    const _Float16* v3p = vtw + (3 * 16 + lk) * VT_LD + koff;
    v16h vf0 = frag_from_f16(v0p, v0p + 16);
    v16h vf1 = frag_from_f16(v1p, v1p + 16);
    v16h vf2 = frag_from_f16(v2p, v2p + 16);
    v16h vf3 = frag_from_f16(v3p, v3p + 16);

    o0 = __builtin_amdgcn_wmma_f32_16x16x32_f16(false, pa, false, vf0, (short)0,
                                                o0, false, false);
    o1 = __builtin_amdgcn_wmma_f32_16x16x32_f16(false, pa, false, vf1, (short)0,
                                                o1, false, false);
    o2 = __builtin_amdgcn_wmma_f32_16x16x32_f16(false, pa, false, vf2, (short)0,
                                                o2, false, false);
    o3 = __builtin_amdgcn_wmma_f32_16x16x32_f16(false, pa, false, vf3, (short)0,
                                                o3, false, false);
  }

  // ---- Cross-wave reduction in LDS (reuse this wave's Vt region as f32
  // partial buffer; 16x64 f32 = 4KB fits in the 5KB region). ----
  {
    float* pred = (float*)(Vt + wave * (64 * VT_LD));
#pragma unroll
    for (int r = 0; r < 8; ++r) {
      pred[(rhi + r) * 64 + 0 + lk]  = o0[r];
      pred[(rhi + r) * 64 + 16 + lk] = o1[r];
      pred[(rhi + r) * 64 + 32 + lk] = o2[r];
      pred[(rhi + r) * 64 + 48 + lk] = o3[r];
    }
    __syncthreads();
    for (int e = tid; e < 16 * 64; e += 256) {
      int row = e >> 6, d = e & 63;
      float s = 0.f;
#pragma unroll
      for (int w = 0; w < 8; ++w)
        s += ((const float*)(Vt + w * (64 * VT_LD)))[row * 64 + d];
      attn_buf[((size_t)(t0 + row) * 8 + b) * 1024 + h * 64 + d] = s;
    }
  }
}

// ---------------------------------------------------------------------------
// Host-side launch orchestration (graph-capture safe: async ops on stream only)
// ---------------------------------------------------------------------------
extern "C" void kernel_launch(void* const* d_in, const int* in_sizes, int n_in,
                              void* d_out, int out_size, void* d_ws,
                              size_t ws_size, hipStream_t stream) {
  (void)in_sizes; (void)n_in; (void)out_size; (void)ws_size;

  const float* audio = (const float*)d_in[0];
  const float* vision = (const float*)d_in[1];
  const float* text = (const float*)d_in[2];
  const float* W_a = (const float*)d_in[3];
  const float* b_a = (const float*)d_in[4];
  const float* W_v = (const float*)d_in[5];
  const float* b_v = (const float*)d_in[6];
  const float* W_l = (const float*)d_in[7];
  const float* b_l = (const float*)d_in[8];
  const float* Wo_a = (const float*)d_in[9];
  const float* bo_a = (const float*)d_in[10];
  const float* Wo_v = (const float*)d_in[11];
  const float* bo_v = (const float*)d_in[12];
  const float* Wo_l = (const float*)d_in[13];
  const float* bo_l = (const float*)d_in[14];
  const unsigned char* mask_a = (const unsigned char*)d_in[15];
  const unsigned char* mask_v = (const unsigned char*)d_in[16];
  const unsigned char* mask_l = (const unsigned char*)d_in[17];

  // Workspace layout (floats)
  float* ws = (float*)d_ws;
  float* qkv_a = ws;                                  // 4096*3072
  float* qkv_v = qkv_a + (size_t)4096 * 3072;         // 4096*3072
  float* qkv_l = qkv_v + (size_t)4096 * 3072;         // 1024*3072
  float* abuf_a = qkv_l + (size_t)1024 * 3072;        // 4096*1024
  float* abuf_v = abuf_a + (size_t)4096 * 1024;       // 4096*1024
  float* abuf_l = abuf_v + (size_t)4096 * 1024;       // 1024*1024
  float* kscale = abuf_l + (size_t)1024 * 1024;       // 8*1152
  float* kbias = kscale + 8 * 1152;                   // 8*1152

  // Output layout (floats): attn_a, attn_v, attn_l, map_a, map_v, map_l
  float* out = (float*)d_out;
  float* out_attn_a = out;                        // 512*8*1024
  float* out_attn_v = out + (size_t)4194304;      // 512*8*1024
  float* out_attn_l = out + (size_t)8388608;      // 128*8*1024
  float* out_map_a = out + (size_t)9437184;       // 8*512*1152
  float* out_map_v = out + (size_t)14155776;      // 8*512*1152
  float* out_map_l = out + (size_t)18874368;      // 8*128*1152

  // Zero the atomically-accumulated map region (contiguous across modalities).
  hipMemsetAsync(out_map_a, 0,
                 ((size_t)4718592 * 2 + 1179648) * sizeof(float), stream);

  init_scales<<<1, 256, 0, stream>>>(mask_a, mask_v, mask_l, kscale, kbias);

  // QKV projections: (L*B, 1024) x (3072, 1024)^T
  wmma_gemm_bias<<<dim3(24, 32), 256, 0, stream>>>(audio, W_a, b_a, qkv_a,
                                                   4096, 3072, 1024);
  wmma_gemm_bias<<<dim3(24, 32), 256, 0, stream>>>(vision, W_v, b_v, qkv_v,
                                                   4096, 3072, 1024);
  wmma_gemm_bias<<<dim3(24, 8), 256, 0, stream>>>(text, W_l, b_l, qkv_l, 1024,
                                                  3072, 1024);

  constexpr size_t ATTN_SMEM =
      (size_t)16 * SC_STRIDE * 4 + (size_t)8 * 64 * VT_LD * 2 + (256 + 32) * 4;

  // Cross-modal attention per query modality
  mhca_attn<<<dim3(32, 128), 256, ATTN_SMEM, stream>>>(
      qkv_a, qkv_a, qkv_v, qkv_l, kscale, kbias, abuf_a, out_map_a, 512);
  mhca_attn<<<dim3(32, 128), 256, ATTN_SMEM, stream>>>(
      qkv_v, qkv_a, qkv_v, qkv_l, kscale, kbias, abuf_v, out_map_v, 512);
  mhca_attn<<<dim3(8, 128), 256, ATTN_SMEM, stream>>>(
      qkv_l, qkv_a, qkv_v, qkv_l, kscale, kbias, abuf_l, out_map_l, 128);

  // Output projections: (T*B, 1024) x (1024, 1024)^T  -> final outputs
  wmma_gemm_bias<<<dim3(8, 32), 256, 0, stream>>>(abuf_a, Wo_a, bo_a,
                                                  out_attn_a, 4096, 1024, 1024);
  wmma_gemm_bias<<<dim3(8, 32), 256, 0, stream>>>(abuf_v, Wo_v, bo_v,
                                                  out_attn_v, 4096, 1024, 1024);
  wmma_gemm_bias<<<dim3(8, 8), 256, 0, stream>>>(abuf_l, Wo_l, bo_l,
                                                 out_attn_l, 1024, 1024, 1024);
}
